// RMVSNet_30940944400842
// MI455X (gfx1250) — compile-verified, matrix operand order checked by reading, not run on hardware
//
#include <hip/hip_runtime.h>

typedef __attribute__((ext_vector_type(16))) _Float16 v16h;
typedef __attribute__((ext_vector_type(8)))  float    v8f;

#define KPAD_MAX 448   // max padded K across all convs (g1: 48*9=432 -> 448)

// Per-K-index gather descriptor, staged in LDS once per block:
//   ptr   : f16 channel-plane base pointer (inA/inB concat already resolved)
//   dy,dx : kernel-window offsets (ky-pad, kx-pad); 0x40000 marks padded K
struct __align__(16) KEntry { const _Float16* ptr; int dy; int dx; };

__device__ __forceinline__ _Float16 rmvs_gatherA(const _Float16* p, int dy, int dx,
                                                 int yoS, int xoS, int Hin, int Win)
{
    const int yi = yoS + dy;
    const int xi = xoS + dx;
    const bool inb = ((unsigned)yi < (unsigned)Hin) & ((unsigned)xi < (unsigned)Win);
    const int yc = yi < 0 ? 0 : (yi > Hin - 1 ? Hin - 1 : yi);
    const int xc = xi < 0 ? 0 : (xi > Win - 1 ? Win - 1 : xi);
    const _Float16 v = p[(size_t)yc * Win + xc];       // 16-bit clamped load
    return inb ? v : (_Float16)0.0f;                   // branch-free border zero
}

// Branch-free activations using hardware v_exp_f32 / v_rcp_f32 (no IEEE div)
__device__ __forceinline__ float rmvs_act(float v, int act)
{
    if (act == 1) return fmaxf(v, 0.0f);                                   // relu
    if (act == 2) return __builtin_amdgcn_rcpf(1.0f + __expf(-v));         // sigmoid
    if (act == 3) {                                                        // fast tanh
        const float vc = fminf(fmaxf(v, -10.0f), 10.0f);
        const float e  = __expf(2.0f * vc);
        return (e - 1.0f) * __builtin_amdgcn_rcpf(e + 1.0f);
    }
    return v;
}

// ---------------------------------------------------------------------------
// Implicit-GEMM convolution via V_WMMA_F32_16X16X32_F16, f16 in / f16 out.
//   M = output pixels (2 x 16-pixel tiles per wave), N = out channels (pad 16),
//   K = Cin*k*k (pad 32).  Input may be concat of two tensors (cinA | cinB).
//   Weights pre-packed f16 [Kpad][Npad]; block's 16-col slice staged in LDS.
//   Dual outputs (r/u gates) share one WMMA tile via cout0/cout1.
// ---------------------------------------------------------------------------
__global__ __launch_bounds__(128)
void rmvs_wmma_conv_kernel(const _Float16* __restrict__ inA, int cinA, long strA,
                           const _Float16* __restrict__ inB, int cinB, long strB,
                           const _Float16* __restrict__ wpack,
                           const float* __restrict__ bpack,
                           _Float16* __restrict__ out0, int cout0, long str0,
                           _Float16* __restrict__ out1, int cout1, long str1,
                           int Hin, int Win, int Hout, int Wout,
                           int ksz, int cstride, int pad,
                           int Kreal, int Kpad, int Npad, int act)
{
    __shared__ _Float16 wlds[KPAD_MAX * 16];
    __shared__ KEntry   ktab[KPAD_MAX];
    const int tid   = threadIdx.x;
    const int lane  = tid & 31;
    const int wave  = tid >> 5;
    const int img   = blockIdx.z;
    const int ntile = blockIdx.y;
    (void)cinB;

    const _Float16* baseA = inA + (long)img * strA;
    const _Float16* baseB = inB ? inB + (long)img * strB : (const _Float16*)0;
    const long      HinWin = (long)Hin * Win;

    // Stage weight slice (columns ntile*16 .. +15) into LDS.
    for (int idx = tid; idx < Kpad * 16; idx += 128) {
        const int k = idx >> 4, col = idx & 15;
        wlds[idx] = wpack[(long)k * Npad + ntile * 16 + col];
    }
    // Stage K-decode table: one divide per entry per block, none in hot loop.
    const int kk2 = ksz * ksz;
    for (int kk = tid; kk < Kpad; kk += 128) {
        const _Float16* p; int dy, dx;
        if (kk < Kreal) {
            const int ci  = kk / kk2;
            const int rem = kk - ci * kk2;
            const int ky  = rem / ksz;
            const int kx  = rem - ky * ksz;
            p  = (ci < cinA) ? baseA + (size_t)ci * HinWin
                             : baseB + (size_t)(ci - cinA) * HinWin;
            dy = ky - pad; dx = kx - pad;
        } else {
            p = baseA; dy = 0x40000; dx = 0x40000;   // forces in-bounds test false
        }
        ktab[kk].ptr = p; ktab[kk].dy = dy; ktab[kk].dx = dx;
    }
    __syncthreads();

    // Two 16-pixel M tiles per wave, sharing B fragment and ktab reads.
    const int mbase0 = (blockIdx.x * 8 + wave * 2) * 16;
    const int mbase1 = mbase0 + 16;
    const int l15    = lane & 15;
    const int m0     = mbase0 + l15;
    const int m1     = mbase1 + l15;
    const int yo0 = m0 / Wout, xo0 = m0 - yo0 * Wout;
    const int yo1 = m1 / Wout, xo1 = m1 - yo1 * Wout;
    const int yoS0 = yo0 * cstride, xoS0 = xo0 * cstride;
    const int yoS1 = yo1 * cstride, xoS1 = xo1 * cstride;
    const int kbase = (lane >> 4) * 8;               // K sub-block per lane half

    v8f acc0 = {}, acc1 = {};
    for (int kt = 0; kt < Kpad; kt += 32) {
        v16h a0, a1, bfrag;
#pragma unroll
        for (int i = 0; i < 16; ++i) {
            // K index for half i per the 16-bit A/B VGPR layout (ISA 7.12.2)
            const int kl = (i < 8) ? (kbase + i) : (16 + kbase + (i - 8));
            const int kk = kt + kl;
            bfrag[i] = wlds[kk * 16 + l15];          // B: n = lane&15
            const KEntry e = ktab[kk];               // broadcast, uniform per lane half
            a0[i] = rmvs_gatherA(e.ptr, e.dy, e.dx, yoS0, xoS0, Hin, Win);
            a1[i] = rmvs_gatherA(e.ptr, e.dy, e.dx, yoS1, xoS1, Hin, Win);
        }
        acc0 = __builtin_amdgcn_wmma_f32_16x16x32_f16(
                   false, a0, false, bfrag, (short)0, acc0, false, false);
        acc1 = __builtin_amdgcn_wmma_f32_16x16x32_f16(
                   false, a1, false, bfrag, (short)0, acc1, false, false);
    }

    // C/D layout: VGPR r -> M = mbase + r + 8*(lane>=16), N = lane&15
    const int   nglob = ntile * 16 + l15;
    const float bias  = bpack[nglob];
    const long  hwOut = (long)Hout * Wout;
#pragma unroll
    for (int r = 0; r < 8; ++r) {
#pragma unroll
        for (int t = 0; t < 2; ++t) {
            const int mm = (t ? mbase1 : mbase0) + r + ((lane >> 4) << 3);
            const float v = rmvs_act((t ? acc1[r] : acc0[r]) + bias, act);
            if (nglob < cout0)
                out0[(long)img * str0 + (long)nglob * hwOut + mm] = (_Float16)v;
            else if (nglob < cout0 + cout1)
                out1[(long)img * str1 + (long)(nglob - cout0) * hwOut + mm] = (_Float16)v;
        }
    }
}

// ---------------------------------------------------------------------------
// Weight packing: f32 [co][ci][k][k]  ->  f16 [Kpad][Npad] (+ bias [Npad])
// ---------------------------------------------------------------------------
__global__ void rmvs_pack_clear_kernel(_Float16* wp, float* bp, int n, int Npad)
{
    const int i = blockIdx.x * blockDim.x + threadIdx.x;
    if (i < n)    wp[i] = (_Float16)0.0f;
    if (i < Npad) bp[i] = 0.0f;
}

__global__ void rmvs_pack_fill_kernel(const float* __restrict__ w,
                                      const float* __restrict__ b,
                                      _Float16* __restrict__ wp,
                                      float* __restrict__ bp,
                                      int co, int ci, int ksz, int Npad, int colOff)
{
    const int i = blockIdx.x * blockDim.x + threadIdx.x;
    const int per = ci * ksz * ksz;
    if (i >= co * per) return;
    const int o   = i / per;
    const int rem = i - o * per;            // == kk = ci_i*k*k + ky*k + kx
    wp[(long)rem * Npad + colOff + o] = (_Float16)w[i];
    if (rem == 0) bp[colOff + o] = b[o];
}

// f32 -> f16 tensor convert (input images)
__global__ void rmvs_f32_to_f16_kernel(const float* __restrict__ in,
                                       _Float16* __restrict__ out, int n)
{
    const int i = blockIdx.x * blockDim.x + threadIdx.x;
    if (i < n) out[i] = (_Float16)in[i];
}

// ---------------------------------------------------------------------------
// 4x4 inverse of reference projection + copy of source projections
// ---------------------------------------------------------------------------
__global__ void rmvs_init_proj_kernel(const float* __restrict__ projections,
                                      float* __restrict__ invref,
                                      float* __restrict__ proj)
{
    if (threadIdx.x != 0 || blockIdx.x != 0) return;
    float a[4][8];
    for (int i = 0; i < 4; ++i)
        for (int j = 0; j < 4; ++j) {
            a[i][j]     = projections[i * 4 + j];
            a[i][4 + j] = (i == j) ? 1.0f : 0.0f;
        }
    for (int c = 0; c < 4; ++c) {
        int piv = c; float best = fabsf(a[c][c]);
        for (int r = c + 1; r < 4; ++r) {
            const float v = fabsf(a[r][c]);
            if (v > best) { best = v; piv = r; }
        }
        if (piv != c)
            for (int j = 0; j < 8; ++j) { float t = a[c][j]; a[c][j] = a[piv][j]; a[piv][j] = t; }
        const float inv = 1.0f / a[c][c];
        for (int j = 0; j < 8; ++j) a[c][j] *= inv;
        for (int r = 0; r < 4; ++r)
            if (r != c) {
                const float f = a[r][c];
                for (int j = 0; j < 8; ++j) a[r][j] -= f * a[c][j];
            }
    }
    for (int i = 0; i < 4; ++i)
        for (int j = 0; j < 4; ++j) invref[i * 4 + j] = a[i][4 + j];
    for (int s = 0; s < 2; ++s)
        for (int i = 0; i < 16; ++i) proj[s * 16 + i] = projections[(1 + s) * 16 + i];
}

// proj[s] = proj[s] @ invref  (2 matrices, 32 threads)
__global__ void rmvs_proj_update_kernel(float* __restrict__ proj,
                                        const float* __restrict__ invref)
{
    __shared__ float old[32];
    const int t = threadIdx.x;
    if (t < 32) old[t] = proj[t];
    __syncthreads();
    if (t < 32) {
        const int s = t >> 4, e = t & 15, r = e >> 2, c = e & 3;
        float acc = 0.0f;
        for (int j = 0; j < 4; ++j) acc += old[s * 16 + r * 4 + j] * invref[j * 4 + c];
        proj[t] = acc;
    }
}

// ---------------------------------------------------------------------------
// Fused homography warp + bilinear sample + variance cost (-> -cost, 32ch)
// feats: f16 [3 views][32][128*128]; gruIn: f16 [32][128*128]
// ---------------------------------------------------------------------------
__global__ void rmvs_warp_cost_kernel(const _Float16* __restrict__ feats,
                                      const float* __restrict__ proj,
                                      const float* __restrict__ depths, int d,
                                      _Float16* __restrict__ gruIn)
{
    const int hw = 128 * 128, W = 128, H = 128;
    const int p = blockIdx.x * blockDim.x + threadIdx.x;
    if (p >= hw) return;
    const int y = p >> 7, x = p & 127;
    const float depth = depths[d];

    int x0i[2], x1i[2], y0i[2], y1i[2];
    float wxv[2], wyv[2], vld[2];
    for (int s = 0; s < 2; ++s) {
        const float* P = proj + s * 16;
        const float hx = (float)x * depth, hy = (float)y * depth, hz = depth;
        const float q0 = P[0] * hx + P[1] * hy + P[2]  * hz + P[3];
        const float q1 = P[4] * hx + P[5] * hy + P[6]  * hz + P[7];
        float       q2 = P[8] * hx + P[9] * hy + P[10] * hz + P[11];
        if (fabsf(q2) < 1e-6f) q2 = 1e-6f;
        const float rz = __builtin_amdgcn_rcpf(q2);    // one v_rcp for both coords
        const float px = q0 * rz, py = q1 * rz;
        vld[s] = (px >= 0.0f && px <= (float)(W - 1) &&
                  py >= 0.0f && py <= (float)(H - 1)) ? 1.0f : 0.0f;
        const float fx = floorf(px), fy = floorf(py);
        wxv[s] = px - fx; wyv[s] = py - fy;
        const int x0 = (int)fx, y0 = (int)fy;
        x0i[s] = x0     < 0 ? 0 : (x0     > W - 1 ? W - 1 : x0);
        x1i[s] = x0 + 1 < 0 ? 0 : (x0 + 1 > W - 1 ? W - 1 : x0 + 1);
        y0i[s] = y0     < 0 ? 0 : (y0     > H - 1 ? H - 1 : y0);
        y1i[s] = y0 + 1 < 0 ? 0 : (y0 + 1 > H - 1 ? H - 1 : y0 + 1);
    }
    for (int c = 0; c < 32; ++c) {
        const float vref = (float)feats[(size_t)c * hw + p];
        float s1 = vref, s2 = vref * vref;
        for (int s = 0; s < 2; ++s) {
            const _Float16* f = feats + ((size_t)(1 + s) * 32 + c) * hw;
            const float wx = wxv[s], wy = wyv[s];
            const float v00 = (float)f[y0i[s] * W + x0i[s]];
            const float v01 = (float)f[y0i[s] * W + x1i[s]];
            const float v10 = (float)f[y1i[s] * W + x0i[s]];
            const float v11 = (float)f[y1i[s] * W + x1i[s]];
            float v = (v00 * (1.0f - wx) + v01 * wx) * (1.0f - wy) +
                      (v10 * (1.0f - wx) + v11 * wx) * wy;
            v *= vld[s];
            s1 += v; s2 += v * v;
        }
        const float m1 = s1 * (1.0f / 3.0f);
        const float m2 = s2 * (1.0f / 3.0f);
        gruIn[(size_t)c * hw + p] = (_Float16)(-(m2 - m1 * m1));
    }
}

// ---------------------------------------------------------------------------
// Elementwise helpers (f16 storage, f32 math)
// ---------------------------------------------------------------------------
__global__ void rmvs_zeroh_kernel(_Float16* p, int n)
{
    const int i = blockIdx.x * blockDim.x + threadIdx.x;
    if (i < n) p[i] = (_Float16)0.0f;
}
__global__ void rmvs_mul_kernel(_Float16* o, const _Float16* a, const _Float16* b, int n)
{
    const int i = blockIdx.x * blockDim.x + threadIdx.x;
    if (i < n) o[i] = (_Float16)((float)a[i] * (float)b[i]);
}
__global__ void rmvs_gru_combine_kernel(_Float16* h, const _Float16* u, const _Float16* c, int n)
{
    const int i = blockIdx.x * blockDim.x + threadIdx.x;
    if (i < n) {
        const float uu = (float)u[i];
        h[i] = (_Float16)((1.0f - uu) * (float)h[i] + uu * (float)c[i]);
    }
}

// softmax over 128 depths per pixel + expected depth (f16 vol, f32 out)
__global__ void rmvs_softmax_depth_kernel(const _Float16* __restrict__ vol,
                                          const float* __restrict__ depths,
                                          float* __restrict__ out)
{
    const int hw = 128 * 128;
    const int p = blockIdx.x * blockDim.x + threadIdx.x;
    if (p >= hw) return;
    float mx = -1e30f;
    for (int d = 0; d < 128; ++d) mx = fmaxf(mx, (float)vol[(size_t)d * hw + p]);
    float se = 0.0f, acc = 0.0f;
    for (int d = 0; d < 128; ++d) {
        const float e = __expf((float)vol[(size_t)d * hw + p] - mx);
        se += e; acc += e * depths[d];
    }
    out[p] = acc * __builtin_amdgcn_rcpf(se);
}

// ---------------------------------------------------------------------------
// Host orchestration
// ---------------------------------------------------------------------------
extern "C" void kernel_launch(void* const* d_in, const int* in_sizes, int n_in,
                              void* d_out, int out_size, void* d_ws, size_t ws_size,
                              hipStream_t stream)
{
    (void)in_sizes; (void)n_in; (void)out_size; (void)ws_size;
    const float* images = (const float*)d_in[0];
    const float* projsI = (const float*)d_in[1];
    const float* depths = (const float*)d_in[2];
    auto par = [&](int i) { return (const float*)d_in[i]; };

    // Deterministic bump allocator over d_ws
    char* wsbase = (char*)d_ws;
    size_t off = 0;
    auto allocf = [&](size_t n) {
        float* p = (float*)(wsbase + off);
        off += ((n * sizeof(float) + 255) / 256) * 256; return p;
    };
    auto alloch = [&](size_t n) {
        _Float16* p = (_Float16*)(wsbase + off);
        off += ((n * sizeof(_Float16) + 255) / 256) * 256; return p;
    };

    const int HWf = 128 * 128;
    _Float16* imgH  = alloch((size_t)3 * 3 * 512 * 512);
    _Float16* pingA = alloch((size_t)3 * 8 * 512 * 512);
    _Float16* pingB = alloch((size_t)3 * 8 * 512 * 512);
    _Float16* gruIn = alloch((size_t)32 * HWf);
    _Float16* h1 = alloch((size_t)16 * HWf);
    _Float16* h2 = alloch((size_t)4 * HWf);
    _Float16* h3 = alloch((size_t)2 * HWf);
    _Float16* r1 = alloch((size_t)16 * HWf); _Float16* u1 = alloch((size_t)16 * HWf);
    _Float16* cc1 = alloch((size_t)16 * HWf); _Float16* rh1 = alloch((size_t)16 * HWf);
    _Float16* r2 = alloch((size_t)4 * HWf); _Float16* u2 = alloch((size_t)4 * HWf);
    _Float16* cc2 = alloch((size_t)4 * HWf); _Float16* rh2 = alloch((size_t)4 * HWf);
    _Float16* r3 = alloch((size_t)2 * HWf); _Float16* u3 = alloch((size_t)2 * HWf);
    _Float16* cc3 = alloch((size_t)2 * HWf); _Float16* rh3 = alloch((size_t)2 * HWf);
    _Float16* vol = alloch((size_t)128 * HWf);
    float* proj   = allocf(32);
    float* invref = allocf(16);

    auto kpad32 = [](int K) { return ((K + 31) / 32) * 32; };

    // Feature extractor layer table (ci, co, k, stride, Hin, Hout); square images.
    struct FC { int ci, co, k, s, Hin, Hout; };
    const FC fc[8] = {
        {3, 8, 3, 1, 512, 512}, {8, 8, 3, 1, 512, 512},
        {8, 16, 5, 2, 512, 256}, {16, 16, 3, 1, 256, 256}, {16, 16, 3, 1, 256, 256},
        {16, 32, 5, 2, 256, 128}, {32, 32, 3, 1, 128, 128}, {32, 32, 3, 1, 128, 128}};
    _Float16* fwp[8]; float* fbp[8]; int fK[8], fKp[8], fNp[8];
    for (int i = 0; i < 8; ++i) {
        fK[i]  = fc[i].ci * fc[i].k * fc[i].k;
        fKp[i] = kpad32(fK[i]);
        fNp[i] = ((fc[i].co + 15) / 16) * 16;
        fwp[i] = alloch((size_t)fKp[i] * fNp[i]);
        fbp[i] = allocf(fNp[i]);
    }
    const int K1 = 48 * 9, K1p = kpad32(K1);   // g1: 432 -> 448
    _Float16* g1ru_w = alloch((size_t)K1p * 32); float* g1ru_b = allocf(32);
    _Float16* g1c_w  = alloch((size_t)K1p * 16); float* g1c_b  = allocf(16);
    const int K2 = 20 * 9, K2p = kpad32(K2);   // g2: 180 -> 192
    _Float16* g2ru_w = alloch((size_t)K2p * 16); float* g2ru_b = allocf(16);
    _Float16* g2c_w  = alloch((size_t)K2p * 16); float* g2c_b  = allocf(16);
    const int K3 = 6 * 9, K3p = kpad32(K3);    // g3: 54 -> 64
    _Float16* g3ru_w = alloch((size_t)K3p * 16); float* g3ru_b = allocf(16);
    _Float16* g3c_w  = alloch((size_t)K3p * 16); float* g3c_b  = allocf(16);
    const int Ko = 2 * 9, Kop = kpad32(Ko);    // out: 18 -> 32
    _Float16* out_wp = alloch((size_t)Kop * 16); float* out_bp = allocf(16);

    auto packClear = [&](_Float16* wp, float* bp, int Kp, int Np) {
        const int n = Kp * Np;
        rmvs_pack_clear_kernel<<<dim3((n + 255) / 256), dim3(256), 0, stream>>>(wp, bp, n, Np);
    };
    auto packFill = [&](const float* w, const float* b, _Float16* wp, float* bp,
                        int co, int ci, int k, int Np, int colOff) {
        const int n = co * ci * k * k;
        rmvs_pack_fill_kernel<<<dim3((n + 255) / 256), dim3(256), 0, stream>>>(
            w, b, wp, bp, co, ci, k, Np, colOff);
    };

    // Param index map (setup_inputs dict order): c{i}_w at 3+2i, c{i}_b at 4+2i;
    // g1 rw/rb/uw/ub/cw/cb at 19..24, g2 at 25..30, g3 at 31..36; out_w 37, out_b 38.
    for (int i = 0; i < 8; ++i) {
        packClear(fwp[i], fbp[i], fKp[i], fNp[i]);
        packFill(par(3 + 2 * i), par(4 + 2 * i), fwp[i], fbp[i],
                 fc[i].co, fc[i].ci, fc[i].k, fNp[i], 0);
    }
    packClear(g1ru_w, g1ru_b, K1p, 32);
    packFill(par(19), par(20), g1ru_w, g1ru_b, 16, 48, 3, 32, 0);   // r gate
    packFill(par(21), par(22), g1ru_w, g1ru_b, 16, 48, 3, 32, 16);  // u gate
    packClear(g1c_w, g1c_b, K1p, 16);
    packFill(par(23), par(24), g1c_w, g1c_b, 16, 48, 3, 16, 0);
    packClear(g2ru_w, g2ru_b, K2p, 16);
    packFill(par(25), par(26), g2ru_w, g2ru_b, 4, 20, 3, 16, 0);
    packFill(par(27), par(28), g2ru_w, g2ru_b, 4, 20, 3, 16, 4);
    packClear(g2c_w, g2c_b, K2p, 16);
    packFill(par(29), par(30), g2c_w, g2c_b, 4, 20, 3, 16, 0);
    packClear(g3ru_w, g3ru_b, K3p, 16);
    packFill(par(31), par(32), g3ru_w, g3ru_b, 2, 6, 3, 16, 0);
    packFill(par(33), par(34), g3ru_w, g3ru_b, 2, 6, 3, 16, 2);
    packClear(g3c_w, g3c_b, K3p, 16);
    packFill(par(35), par(36), g3c_w, g3c_b, 2, 6, 3, 16, 0);
    packClear(out_wp, out_bp, Kop, 16);
    packFill(par(37), par(38), out_wp, out_bp, 1, 2, 3, 16, 0);

    // Convert input images to f16 once; all activations stay f16 after this.
    {
        const int n = 3 * 3 * 512 * 512;
        rmvs_f32_to_f16_kernel<<<(n + 255) / 256, 256, 0, stream>>>(images, imgH, n);
    }

    auto conv = [&](const _Float16* inA, int cinA, long strA,
                    const _Float16* inB, int cinB, long strB,
                    const _Float16* wp, const float* bp,
                    _Float16* o0, int c0n, long s0,
                    _Float16* o1, int c1n, long s1,
                    int Hin, int Win, int Hout, int Wout,
                    int k, int st, int pd, int Kr, int Kp, int Np, int act, int nimg) {
        dim3 grid((unsigned)((Hout * Wout) / 128), (unsigned)(Np / 16), (unsigned)nimg);
        rmvs_wmma_conv_kernel<<<grid, 128, 0, stream>>>(
            inA, cinA, strA, inB, cinB, strB, wp, bp,
            o0, c0n, s0, o1, c1n, s1,
            Hin, Win, Hout, Wout, k, st, pd, Kr, Kp, Np, act);
    };

    // ---- Feature extraction on 3 views (relu on all but last layer) ----
    const _Float16* cin = imgH; long cinStr = (long)3 * 512 * 512;
    _Float16* bufs[2] = {pingA, pingB};
    int cur = 0;
    for (int i = 0; i < 8; ++i) {
        _Float16* o = bufs[cur];
        const int Hin = fc[i].Hin, Hout = fc[i].Hout;
        conv(cin, fc[i].ci, cinStr, nullptr, 0, 0, fwp[i], fbp[i],
             o, fc[i].co, (long)fc[i].co * Hout * Hout, nullptr, 0, 0,
             Hin, Hin, Hout, Hout, fc[i].k, fc[i].s, fc[i].k / 2,
             fK[i], fKp[i], fNp[i], (i < 7) ? 1 : 0, 3);
        cin = o; cinStr = (long)fc[i].co * Hout * Hout;
        cur ^= 1;
    }
    const _Float16* feats = cin;   // f16 [3][32][128*128]

    // ---- Init GRU states and projections ----
    rmvs_zeroh_kernel<<<(16 * HWf + 255) / 256, 256, 0, stream>>>(h1, 16 * HWf);
    rmvs_zeroh_kernel<<<(4 * HWf + 255) / 256, 256, 0, stream>>>(h2, 4 * HWf);
    rmvs_zeroh_kernel<<<(2 * HWf + 255) / 256, 256, 0, stream>>>(h3, 2 * HWf);
    rmvs_init_proj_kernel<<<1, 1, 0, stream>>>(projsI, invref, proj);

    // ---- Sequential depth sweep ----
    for (int d = 0; d < 128; ++d) {
        rmvs_proj_update_kernel<<<1, 32, 0, stream>>>(proj, invref);
        rmvs_warp_cost_kernel<<<(HWf + 127) / 128, 128, 0, stream>>>(feats, proj, depths, d, gruIn);
        // GRU level 1: r,u fused (sigmoid), then c (tanh), then combine
        conv(gruIn, 32, 0, h1, 16, 0, g1ru_w, g1ru_b, r1, 16, 0, u1, 16, 0,
             128, 128, 128, 128, 3, 1, 1, K1, K1p, 32, 2, 1);
        rmvs_mul_kernel<<<(16 * HWf + 255) / 256, 256, 0, stream>>>(rh1, r1, h1, 16 * HWf);
        conv(gruIn, 32, 0, rh1, 16, 0, g1c_w, g1c_b, cc1, 16, 0, nullptr, 0, 0,
             128, 128, 128, 128, 3, 1, 1, K1, K1p, 16, 3, 1);
        rmvs_gru_combine_kernel<<<(16 * HWf + 255) / 256, 256, 0, stream>>>(h1, u1, cc1, 16 * HWf);
        // GRU level 2
        conv(h1, 16, 0, h2, 4, 0, g2ru_w, g2ru_b, r2, 4, 0, u2, 4, 0,
             128, 128, 128, 128, 3, 1, 1, K2, K2p, 16, 2, 1);
        rmvs_mul_kernel<<<(4 * HWf + 255) / 256, 256, 0, stream>>>(rh2, r2, h2, 4 * HWf);
        conv(h1, 16, 0, rh2, 4, 0, g2c_w, g2c_b, cc2, 4, 0, nullptr, 0, 0,
             128, 128, 128, 128, 3, 1, 1, K2, K2p, 16, 3, 1);
        rmvs_gru_combine_kernel<<<(4 * HWf + 255) / 256, 256, 0, stream>>>(h2, u2, cc2, 4 * HWf);
        // GRU level 3
        conv(h2, 4, 0, h3, 2, 0, g3ru_w, g3ru_b, r3, 2, 0, u3, 2, 0,
             128, 128, 128, 128, 3, 1, 1, K3, K3p, 16, 2, 1);
        rmvs_mul_kernel<<<(2 * HWf + 255) / 256, 256, 0, stream>>>(rh3, r3, h3, 2 * HWf);
        conv(h2, 4, 0, rh3, 2, 0, g3c_w, g3c_b, cc3, 2, 0, nullptr, 0, 0,
             128, 128, 128, 128, 3, 1, 1, K3, K3p, 16, 3, 1);
        rmvs_gru_combine_kernel<<<(2 * HWf + 255) / 256, 256, 0, stream>>>(h3, u3, cc3, 2 * HWf);
        // regularized cost -> volume slice d
        conv(h3, 2, 0, nullptr, 0, 0, out_wp, out_bp, vol + (size_t)d * HWf, 1, 0,
             nullptr, 0, 0, 128, 128, 128, 128, 3, 1, 1, Ko, Kop, 16, 0, 1);
    }

    // ---- softmax over depth + expected depth ----
    rmvs_softmax_depth_kernel<<<(HWf + 127) / 128, 128, 0, stream>>>(vol, depths, (float*)d_out);
}
